// Pointnet2FeatureAbstract_64201171140769
// MI455X (gfx1250) — matrix operand order repaired and checked by reading, not compiled, not run on hardware
//
#include <hip/hip_runtime.h>
#include <math.h>

// ---------------------------------------------------------------------------
// PointNet++ MSG forward for gfx1250 (MI455X).
// All MLP / conv1 GEMMs run on v_wmma_f32_16x16x32_bf16 (f32 accumulate),
// 2x2 tiles (32Mx32N) per wave for operand reuse: 16 b128 loads -> 4 WMMAs.
// GEMM operands are K-padded to multiples of 32: no divergence in the loop.
// ---------------------------------------------------------------------------

typedef __bf16 bf16_t;
typedef bf16_t v16bf __attribute__((ext_vector_type(16)));
typedef float  v8f   __attribute__((ext_vector_type(8)));
typedef float  v4f   __attribute__((ext_vector_type(4)));

#define EPS_BN_C 1e-5f
#define EPS_GN_C 1e-5f

static inline int rnd32(int x) { return (x + 31) & ~31; }

// ---------------- (B,3,N) -> (B,N,3) ----------------
__global__ void k_transpose_xyz(const float* __restrict__ x, float* __restrict__ o,
                                int B, int N) {
  int i = blockIdx.x * blockDim.x + threadIdx.x;
  if (i >= B * N * 3) return;
  int k = i % 3; int n = (i / 3) % N; int b = i / (3 * N);
  o[i] = x[((size_t)b * 3 + k) * N + n];
}

// ---------------- farthest point sampling: one block per batch ----------------
__global__ void k_fps(const float* __restrict__ xyz, int* __restrict__ out,
                      int N, int np) {
  __shared__ float sd[1024];
  __shared__ int   si[1024];
  __shared__ int   sfar;
  int b = blockIdx.x, t = threadIdx.x;
  const float* X = xyz + (size_t)b * N * 3;
  float dloc[4];
  int ppt = (N + 1023) / 1024;
#pragma unroll
  for (int j = 0; j < 4; j++) dloc[j] = 1e10f;
  if (t == 0) sfar = 0;
  __syncthreads();
  for (int it = 0; it < np; ++it) {
    int far = sfar;
    if (t == 0) out[b * np + it] = far;          // record BEFORE update (scan ys)
    float cx = X[far * 3 + 0], cy = X[far * 3 + 1], cz = X[far * 3 + 2];
    float best = -1.0f; int bi = 0;
    for (int j = 0; j < ppt; j++) {
      int p = t + j * 1024;
      if (p < N) {
        float dx = X[p * 3 + 0] - cx, dy = X[p * 3 + 1] - cy, dz = X[p * 3 + 2] - cz;
        float dd = dx * dx + dy * dy + dz * dz;
        float dmin = fminf(dloc[j], dd);
        dloc[j] = dmin;
        if (dmin > best) { best = dmin; bi = p; }
      }
    }
    sd[t] = best; si[t] = bi;
    __syncthreads();
    for (int s = 512; s > 0; s >>= 1) {
      if (t < s) {
        float d2 = sd[t + s]; int j2 = si[t + s];
        if (d2 > sd[t] || (d2 == sd[t] && j2 < si[t])) { sd[t] = d2; si[t] = j2; }
      }
      __syncthreads();
    }
    if (t == 0) sfar = si[0];
    __syncthreads();
  }
}

// ---------------- gather sampled centers ----------------
__global__ void k_gather_xyz(const float* __restrict__ xyz, const int* __restrict__ fidx,
                             float* __restrict__ nx, int B, int N, int np) {
  int i = blockIdx.x * blockDim.x + threadIdx.x;
  if (i >= B * np * 3) return;
  int k = i % 3; int q = (i / 3) % np; int b = i / (3 * np);
  nx[i] = xyz[((size_t)b * N + fidx[b * np + q]) * 3 + k];
}

// ---------------- ball query: first ns indices (ascending) within r^2 ----------------
__global__ void k_ball(const float* __restrict__ xyz, const float* __restrict__ nx,
                       int* __restrict__ gidx, int B, int N, int np, int ns, float r2) {
  int i = blockIdx.x * blockDim.x + threadIdx.x;
  if (i >= B * np) return;
  int b = i / np;
  const float* X = xyz + (size_t)b * N * 3;
  float cx = nx[i * 3 + 0], cy = nx[i * 3 + 1], cz = nx[i * 3 + 2];
  int* g = gidx + (size_t)i * ns;
  int cnt = 0, first = -1;
  for (int n = 0; n < N && cnt < ns; n++) {
    float dx = X[n * 3] - cx, dy = X[n * 3 + 1] - cy, dz = X[n * 3 + 2] - cz;
    if (dx * dx + dy * dy + dz * dz <= r2) {
      if (first < 0) first = n;
      g[cnt++] = n;
    }
  }
  if (first < 0) first = N - 1;
  for (int s = cnt; s < ns; s++) g[s] = first;
}

// ------- build grouped features [pts(gidx), xyz(gidx)-center], K-padded -------
__global__ void k_group(const float* __restrict__ xyz, const float* __restrict__ nx,
                        const float* __restrict__ pts, const int* __restrict__ gidx,
                        float* __restrict__ out, int B, int N, int np, int ns,
                        int cin, int Cs) {
  int row = blockIdx.x * blockDim.x + threadIdx.x;
  if (row >= B * np * ns) return;
  int b = row / (np * ns); int q = (row / ns) % np;
  int g = gidx[row];
  float* o = out + (size_t)row * Cs;
  if (pts) {
    const float* p = pts + ((size_t)b * N + g) * cin;
    for (int c = 0; c < cin; c++) o[c] = p[c];
  }
  const float* xp = xyz + ((size_t)b * N + g) * 3;
  const float* cp = nx + ((size_t)b * np + q) * 3;
  o[cin + 0] = xp[0] - cp[0];
  o[cin + 1] = xp[1] - cp[1];
  o[cin + 2] = xp[2] - cp[2];
  for (int c = cin + 3; c < Cs; c++) o[c] = 0.0f;   // zero K-pad
}

// ---------------- pack weights (cout,cin) -> (cout,Kp) with zero pad ----------------
__global__ void k_pack_w(const float* __restrict__ w, float* __restrict__ o,
                         int N, int K, int Kp) {
  int i = blockIdx.x * blockDim.x + threadIdx.x;
  if (i >= N * Kp) return;
  int k = i % Kp, n = i / Kp;
  o[i] = (k < K) ? w[(size_t)n * K + k] : 0.0f;
}

// ---------------- fragment loaders (f32 -> bf16) ----------------
// A frag: lane L<16 -> row M=L, K={0..15}; lane L+16 -> K={16..31}.
// p already includes row base + half*16.
__device__ inline v16bf ld_afrag(const float* __restrict__ p) {
  v4f x0 = *(const v4f*)(p);
  v4f x1 = *(const v4f*)(p + 4);
  v4f x2 = *(const v4f*)(p + 8);
  v4f x3 = *(const v4f*)(p + 12);
  v16bf f;
#pragma unroll
  for (int i = 0; i < 4; i++) {
    f[i]      = (bf16_t)x0[i];
    f[4 + i]  = (bf16_t)x1[i];
    f[8 + i]  = (bf16_t)x2[i];
    f[12 + i] = (bf16_t)x3[i];
  }
  return f;
}
// B frag: lane -> column N=lane&15, K = (lane>>4)*16 + {0..15} (== one Wp row).
// p already includes row base + half*16.
__device__ inline v16bf ld_bfrag(const float* __restrict__ p) {
  v4f x0 = *(const v4f*)(p);
  v4f x1 = *(const v4f*)(p + 4);
  v4f x2 = *(const v4f*)(p + 8);
  v4f x3 = *(const v4f*)(p + 12);
  v16bf f;
#pragma unroll
  for (int i = 0; i < 4; i++) {
    f[i]      = (bf16_t)x0[i];
    f[4 + i]  = (bf16_t)x1[i];
    f[8 + i]  = (bf16_t)x2[i];
    f[12 + i] = (bf16_t)x3[i];
  }
  return f;
}

// C/D layout: VGPR r -> M = r + 8*(lane>>4), N = lane&15.
__device__ inline void st_tile(float* __restrict__ Y, v8f acc, int mb, int col,
                               int M, int N, int ldy, float s, float bb,
                               int relu, int half) {
#pragma unroll
  for (int r = 0; r < 8; r++) {
    int rr = mb + half * 8 + r;
    if (rr < M) {
      float v = acc[r] * s + bb;
      if (relu) v = fmaxf(v, 0.0f);
      Y[(size_t)rr * ldy + col] = (col < N) ? v : 0.0f;
    }
  }
}

// ---------------- WMMA GEMM: Y = act((A[MxKp] * Wp[NxKp]^T)*s + bias) ----------------
// One wave per 32x32 output tile (2x2 fragments), K-loop in chunks of 32.
// Out-of-range rows/cols are clamped on load (results never stored);
// pad output columns [N,ldy) are written as zero (they are the next layer's K-pad).
__global__ void __launch_bounds__(32) k_gemm_wmma(
    const float* __restrict__ A, const float* __restrict__ Wp,
    const float* __restrict__ bias, const float* __restrict__ gma,
    const float* __restrict__ bta, float* __restrict__ Y,
    int M, int N, int Kp, int ldy, int relu) {
  int lane = threadIdx.x & 31;
  int half = lane >> 4;
  int l16  = lane & 15;
  int m0 = blockIdx.x * 32;
  int n0 = blockIdx.y * 32;
  int am0 = m0 + l16;      if (am0 >= M) am0 = M - 1;
  int am1 = m0 + 16 + l16; if (am1 >= M) am1 = M - 1;
  int bn0 = n0 + l16;      if (bn0 >= N) bn0 = N - 1;
  int bn1 = n0 + 16 + l16; if (bn1 >= N) bn1 = N - 1;
  const float* ap0 = A  + (size_t)am0 * Kp + half * 16;
  const float* ap1 = A  + (size_t)am1 * Kp + half * 16;
  const float* bp0 = Wp + (size_t)bn0 * Kp + half * 16;
  const float* bp1 = Wp + (size_t)bn1 * Kp + half * 16;
  v8f acc00 = {}, acc01 = {}, acc10 = {}, acc11 = {};
  for (int kk = 0; kk < Kp; kk += 32) {
    v16bf a0 = ld_afrag(ap0 + kk);
    v16bf a1 = ld_afrag(ap1 + kk);
    v16bf b0 = ld_bfrag(bp0 + kk);
    v16bf b1 = ld_bfrag(bp1 + kk);
    acc00 = __builtin_amdgcn_wmma_f32_16x16x32_bf16(
        false, a0, false, b0, (short)0, acc00, false, false);
    acc01 = __builtin_amdgcn_wmma_f32_16x16x32_bf16(
        false, a0, false, b1, (short)0, acc01, false, false);
    acc10 = __builtin_amdgcn_wmma_f32_16x16x32_bf16(
        false, a1, false, b0, (short)0, acc10, false, false);
    acc11 = __builtin_amdgcn_wmma_f32_16x16x32_bf16(
        false, a1, false, b1, (short)0, acc11, false, false);
  }
  float scl = rsqrtf(1.0f + EPS_BN_C);
  int col0 = n0 + l16;
  int col1 = n0 + 16 + l16;
  int cc0 = (col0 < N) ? col0 : N - 1;
  int cc1 = (col1 < N) ? col1 : N - 1;
  float s0  = gma ? gma[cc0] * scl : 1.0f;
  float s1  = gma ? gma[cc1] * scl : 1.0f;
  float bb0 = (bias ? bias[cc0] * s0 : 0.0f) + (bta ? bta[cc0] : 0.0f);
  float bb1 = (bias ? bias[cc1] * s1 : 0.0f) + (bta ? bta[cc1] : 0.0f);
  st_tile(Y, acc00, m0,      col0, M, N, ldy, s0, bb0, relu, half);
  st_tile(Y, acc01, m0,      col1, M, N, ldy, s1, bb1, relu, half);
  st_tile(Y, acc10, m0 + 16, col0, M, N, ldy, s0, bb0, relu, half);
  st_tile(Y, acc11, m0 + 16, col1, M, N, ldy, s1, bb1, relu, half);
}

// ---------------- max-pool over samples, write into concat slot ----------------
__global__ void k_maxpool(const float* __restrict__ x, float* __restrict__ o,
                          int B, int np, int ns, int C, int ldx, int Ctot, int coff) {
  int i = blockIdx.x * blockDim.x + threadIdx.x;
  if (i >= B * np * C) return;
  int c = i % C; int q = (i / C) % np; int b = i / (C * np);
  const float* xp = x + ((size_t)(b * np + q) * ns) * ldx + c;
  float m = -1e30f;
  for (int s = 0; s < ns; s++) m = fmaxf(m, xp[(size_t)s * ldx]);
  o[(size_t)(b * np + q) * Ctot + coff + c] = m;
}

// ---------------- copy points1 into concat columns [0,C) ----------------
__global__ void k_copy_cols(const float* __restrict__ src, float* __restrict__ dst,
                            int rows, int C, int Ctot, int coff) {
  int i = blockIdx.x * blockDim.x + threadIdx.x;
  if (i >= rows * C) return;
  int c = i % C; int r = i / C;
  dst[(size_t)r * Ctot + coff + c] = src[(size_t)r * C + c];
}

// ---------------- 3-NN inverse-distance interpolation ----------------
__global__ void k_interp(const float* __restrict__ x1, const float* __restrict__ x2,
                         const float* __restrict__ p2, float* __restrict__ dst,
                         int B, int N1, int N2, int C2, int Ctot, int coff) {
  int i = blockIdx.x * blockDim.x + threadIdx.x;
  if (i >= B * N1) return;
  int b = i / N1;
  float cx = x1[i * 3], cy = x1[i * 3 + 1], cz = x1[i * 3 + 2];
  const float* X2 = x2 + (size_t)b * N2 * 3;
  float d0 = 1e30f, d1 = 1e30f, d2 = 1e30f; int i0 = 0, i1 = 0, i2 = 0;
  for (int n = 0; n < N2; n++) {
    float dx = X2[n * 3] - cx, dy = X2[n * 3 + 1] - cy, dz = X2[n * 3 + 2] - cz;
    float d = dx * dx + dy * dy + dz * dz;
    if (d < d0)      { d2 = d1; i2 = i1; d1 = d0; i1 = i0; d0 = d; i0 = n; }
    else if (d < d1) { d2 = d1; i2 = i1; d1 = d;  i1 = n; }
    else if (d < d2) { d2 = d;  i2 = n; }
  }
  float w0 = 1.0f / (d0 + 1e-8f), w1 = 1.0f / (d1 + 1e-8f), w2 = 1.0f / (d2 + 1e-8f);
  float ws = w0 + w1 + w2; w0 /= ws; w1 /= ws; w2 /= ws;
  const float* P = p2 + (size_t)b * N2 * C2;
  float* o = dst + (size_t)i * Ctot + coff;
  for (int c = 0; c < C2; c++)
    o[c] = w0 * P[(size_t)i0 * C2 + c] + w1 * P[(size_t)i1 * C2 + c] + w2 * P[(size_t)i2 * C2 + c];
}

// ---------------- GroupNorm + affine + ReLU, (B,N,C) -> (B,C,N) ----------------
__global__ void k_gn_relu(const float* __restrict__ y, const float* __restrict__ g,
                          const float* __restrict__ bt, float* __restrict__ o,
                          int B, int C, int N, int G) {
  __shared__ float ss[256], sq[256];
  int bg = blockIdx.x; int b = bg / G; int gr = bg % G;
  int cpg = C / G; int cnt = cpg * N;
  int t = threadIdx.x;
  float s = 0.0f, q = 0.0f;
  for (int e = t; e < cnt; e += 256) {
    int c = gr * cpg + e / N; int n = e % N;
    float v = y[((size_t)b * N + n) * C + c];
    s += v; q += v * v;
  }
  ss[t] = s; sq[t] = q; __syncthreads();
  for (int st = 128; st > 0; st >>= 1) {
    if (t < st) { ss[t] += ss[t + st]; sq[t] += sq[t + st]; }
    __syncthreads();
  }
  float mean = ss[0] / (float)cnt;
  float var  = sq[0] / (float)cnt - mean * mean;
  float inv  = rsqrtf(var + EPS_GN_C);
  for (int e = t; e < cnt; e += 256) {
    int c = gr * cpg + e / N; int n = e % N;
    float v = (y[((size_t)b * N + n) * C + c] - mean) * inv;
    v = v * g[c] + bt[c];
    o[((size_t)b * C + c) * N + n] = fmaxf(v, 0.0f);
  }
}

// ---------------------------------------------------------------------------

struct LayerP { const float *w, *b, *g, *beta; int cin, cout; };

static inline dim3 g1(int total) { return dim3((total + 255) / 256); }

extern "C" void kernel_launch(void* const* d_in, const int* in_sizes, int n_in,
                              void* d_out, int out_size, void* d_ws, size_t ws_size,
                              hipStream_t stream) {
  (void)in_sizes; (void)n_in; (void)out_size; (void)ws_size;
  static const int   SA_NP[4]      = {1024, 256, 64, 16};
  static const float SA_R[4][2]    = {{0.1f,0.2f},{0.2f,0.4f},{0.4f,0.8f},{0.8f,1.6f}};
  static const int   SA_NS[2]      = {16, 32};
  static const int   SA_CIN[4]     = {0, 96, 256, 512};
  static const int   SA_W[4][2][3] = {
      {{16,16,32},  {32,32,64}},
      {{64,64,128}, {64,96,128}},
      {{128,196,256},{128,196,256}},
      {{256,256,512},{256,384,512}}};
  static const int SA_CTOT[4] = {96, 256, 512, 1024};
  static const int FP_CIN[4]  = {1536, 512, 352, 128};   // fp4, fp3, fp2, fp1
  static const int FP_NL[4]   = {2, 2, 2, 3};
  static const int FP_W[4][3] = {{256,256,0},{256,256,0},{256,128,0},{128,128,128}};
  const int B = 16, NN = 4096;

  // ---- walk inputs in setup_inputs() insertion order ----
  int ip = 0;
  const float* xyz = (const float*)d_in[ip++];
  LayerP sa[4][2][3];
  for (int l = 0; l < 4; l++)
    for (int br = 0; br < 2; br++) {
      int cin = SA_CIN[l] + 3;
      for (int j = 0; j < 3; j++) {
        LayerP& L = sa[l][br][j];
        L.w = (const float*)d_in[ip++]; L.b    = (const float*)d_in[ip++];
        L.g = (const float*)d_in[ip++]; L.beta = (const float*)d_in[ip++];
        L.cin = cin; L.cout = SA_W[l][br][j]; cin = L.cout;
      }
    }
  LayerP fp[4][3];
  for (int f = 0; f < 4; f++) {
    int cin = FP_CIN[f];
    for (int j = 0; j < FP_NL[f]; j++) {
      LayerP& L = fp[f][j];
      L.w = (const float*)d_in[ip++]; L.b    = (const float*)d_in[ip++];
      L.g = (const float*)d_in[ip++]; L.beta = (const float*)d_in[ip++];
      L.cin = cin; L.cout = FP_W[f][j]; cin = L.cout;
    }
  }
  const float* conv_w = (const float*)d_in[ip++];
  const float* conv_b = (const float*)d_in[ip++];
  const float* gn_g   = (const float*)d_in[ip++];
  const float* gn_bt  = (const float*)d_in[ip++];

  // ---- carve workspace ----
  char* ws = (char*)d_ws; size_t off = 0;
  auto alloc = [&](size_t bytes) -> void* {
    void* p = ws + off; off = (off + bytes + 255) & ~(size_t)255; return p;
  };
  float* l0x = (float*)alloc((size_t)B * NN * 3 * 4);
  float* lx[4];
  for (int l = 0; l < 4; l++) lx[l] = (float*)alloc((size_t)B * SA_NP[l] * 3 * 4);
  int* fidx = (int*)alloc((size_t)B * 1024 * 4);
  int* gidx = (int*)alloc((size_t)B * 1024 * 32 * 4);
  size_t bufN = (size_t)B * 1024 * 32 * 64;          // max rows * max padded width
  float* bufA = (float*)alloc(bufN * 4);
  float* bufB = (float*)alloc(bufN * 4);
  float* wbuf = (float*)alloc((size_t)512 * 1536 * 4);  // packed-weight scratch
  float* lp[4];
  lp[0] = (float*)alloc((size_t)B * 1024 * 96 * 4);
  lp[1] = (float*)alloc((size_t)B * 256 * 256 * 4);
  lp[2] = (float*)alloc((size_t)B * 64 * 512 * 4);
  lp[3] = (float*)alloc((size_t)B * 16 * 1024 * 4);
  float* l3n  = (float*)alloc((size_t)B * 64 * 256 * 4);
  float* l2n  = (float*)alloc((size_t)B * 256 * 256 * 4);
  float* l1n  = (float*)alloc((size_t)B * 1024 * 128 * 4);
  float* l0p  = (float*)alloc((size_t)B * 4096 * 128 * 4);
  float* ybuf = (float*)alloc((size_t)B * 4096 * 128 * 4);

  k_transpose_xyz<<<g1(B * NN * 3), 256, 0, stream>>>(xyz, l0x, B, NN);

  // pack weights, then GEMM; returns the (padded) leading dim of Y.
  auto gemm = [&](const float* Aptr, const LayerP& L, float* Yp, int M, int relu) -> int {
    int Kp  = rnd32(L.cin);
    int ldy = rnd32(L.cout);
    k_pack_w<<<g1(L.cout * Kp), 256, 0, stream>>>(L.w, wbuf, L.cout, L.cin, Kp);
    dim3 grid((M + 31) / 32, ldy / 32);
    k_gemm_wmma<<<grid, 32, 0, stream>>>(Aptr, wbuf, L.b, L.g, L.beta, Yp,
                                         M, L.cout, Kp, ldy, relu);
    return ldy;
  };

  // ---- set abstraction levels ----
  for (int l = 0; l < 4; l++) {
    const float* xin = (l == 0) ? l0x : lx[l - 1];
    int Nin          = (l == 0) ? NN  : SA_NP[l - 1];
    const float* pin = (l == 0) ? nullptr : lp[l - 1];
    int cin = SA_CIN[l];
    int np  = SA_NP[l];
    k_fps<<<B, 1024, 0, stream>>>(xin, fidx, Nin, np);
    k_gather_xyz<<<g1(B * np * 3), 256, 0, stream>>>(xin, fidx, lx[l], B, Nin, np);
    int coff = 0;
    for (int br = 0; br < 2; br++) {
      int ns = SA_NS[br];
      float r = SA_R[l][br];
      k_ball<<<g1(B * np), 256, 0, stream>>>(xin, lx[l], gidx, B, Nin, np, ns, r * r);
      int rows = B * np * ns;
      int Cs = rnd32(cin + 3);
      k_group<<<g1(rows), 256, 0, stream>>>(xin, lx[l], pin, gidx, bufA,
                                            B, Nin, np, ns, cin, Cs);
      float* io[2] = {bufA, bufB}; int cur = 0; int ld = Cs;
      for (int j = 0; j < 3; j++) { ld = gemm(io[cur], sa[l][br][j], io[1 - cur], rows, 1); cur = 1 - cur; }
      int cl = sa[l][br][2].cout;
      k_maxpool<<<g1(B * np * cl), 256, 0, stream>>>(io[cur], lp[l], B, np, ns,
                                                     cl, ld, SA_CTOT[l], coff);
      coff += cl;
    }
  }

  // ---- feature propagation ----
  struct FPd { const float* x1; const float* x2; int N1, N2;
               const float* p1; int C1; const float* p2; int C2; float* dst; };
  FPd fpd[4] = {
      { lx[2], lx[3],   64,   16, lp[2], 512, lp[3], 1024, l3n },
      { lx[1], lx[2],  256,   64, lp[1], 256, l3n,   256,  l2n },
      { lx[0], lx[1], 1024,  256, lp[0],  96, l2n,   256,  l1n },
      { l0x,   lx[0], 4096, 1024, nullptr, 0, l1n,   128,  l0p },
  };
  for (int f = 0; f < 4; f++) {
    FPd& d = fpd[f];
    int rows = B * d.N1; int Ctot = d.C1 + d.C2;   // all multiples of 32
    if (d.p1) k_copy_cols<<<g1(rows * d.C1), 256, 0, stream>>>(d.p1, bufA, rows, d.C1, Ctot, 0);
    k_interp<<<g1(rows), 256, 0, stream>>>(d.x1, d.x2, d.p2, bufA, B, d.N1, d.N2, d.C2, Ctot, d.C1);
    float* io[2] = {bufA, bufB}; int cur = 0;
    int nl = FP_NL[f];
    for (int j = 0; j < nl; j++) {
      float* out = (j == nl - 1) ? d.dst : io[1 - cur];
      gemm(io[cur], fp[f][j], out, rows, 1);
      cur = 1 - cur;
    }
  }

  // ---- conv1 (no BN-affine, no relu) + GroupNorm + ReLU ----
  LayerP cv; cv.w = conv_w; cv.b = conv_b; cv.g = nullptr; cv.beta = nullptr;
  cv.cin = 128; cv.cout = 128;
  gemm(l0p, cv, ybuf, B * 4096, 0);
  k_gn_relu<<<B * 8, 256, 0, stream>>>(ybuf, gn_g, gn_bt, (float*)d_out, B, 128, 4096, 8);
}
